// NCELoss_25460566131136
// MI455X (gfx1250) — matrix operand order repaired and checked by reading, not compiled, not run on hardware
//
#include <hip/hip_runtime.h>
#include <hip/hip_bf16.h>
#include <math.h>

typedef __attribute__((ext_vector_type(16))) _Float16 v16h;
typedef __attribute__((ext_vector_type(8)))  _Float16 v8h;
typedef __attribute__((ext_vector_type(4)))  _Float16 v4h;
typedef __attribute__((ext_vector_type(8)))  float    v8f;
typedef __attribute__((ext_vector_type(4)))  unsigned int uint32x4;
typedef __attribute__((ext_vector_type(8)))  int      int32x8;
typedef __attribute__((ext_vector_type(4)))  int      int32x4;

#define BATCH 128
#define LQ    32
#define LK    128
#define DIM   128
// LDS panel row stride in halves: 128 data + 8 pad (TDM pads 4 dwords per
// 64-dword row) -> row n starts at bank 4n -> conflict-free b128 reads.
#define PSTRIDE 136

// ---------------------------------------------------------------------------
// Kernel 1: L2-normalize rows of left/right (f32) and emit f16 rows.
// ---------------------------------------------------------------------------
__global__ void __launch_bounds__(256)
normalize_to_f16(const float* __restrict__ left,
                 const float* __restrict__ right,
                 _Float16* __restrict__ Lh,
                 _Float16* __restrict__ Rh) {
  const int wave = threadIdx.x >> 5;
  const int lane = threadIdx.x & 31;
  const int row  = blockIdx.x * 8 + wave;   // 0 .. 20479

  const float* src;
  _Float16*    dst;
  if (row < BATCH * LQ) {
    src = left  + (size_t)row * DIM;
    dst = Lh    + (size_t)row * DIM;
  } else {
    const int r = row - BATCH * LQ;
    src = right + (size_t)r * DIM;
    dst = Rh    + (size_t)r * DIM;
  }

  const float4 v = ((const float4*)src)[lane];
  float ss = v.x * v.x + v.y * v.y + v.z * v.z + v.w * v.w;
  ss += __shfl_xor(ss, 1);
  ss += __shfl_xor(ss, 2);
  ss += __shfl_xor(ss, 4);
  ss += __shfl_xor(ss, 8);
  ss += __shfl_xor(ss, 16);
  const float inv = 1.0f / fmaxf(sqrtf(ss), 1e-12f);

  v4h o;
  o[0] = (_Float16)(v.x * inv);
  o[1] = (_Float16)(v.y * inv);
  o[2] = (_Float16)(v.z * inv);
  o[3] = (_Float16)(v.w * inv);
  *(v4h*)(dst + lane * 4) = o;
}

// max across the 16-lane half the calling lane belongs to (xor bits 0..3 only)
__device__ __forceinline__ float halfmax16(float v) {
  v = fmaxf(v, __shfl_xor(v, 1));
  v = fmaxf(v, __shfl_xor(v, 2));
  v = fmaxf(v, __shfl_xor(v, 4));
  v = fmaxf(v, __shfl_xor(v, 8));
  return v;
}

// ---------------------------------------------------------------------------
// Kernel 2: sim[x,y] = sum_i max_j (Lh[x] @ Rh[y]^T)[i,j]
// 8 waves/block share one y; the TDM DMAs the 128x128 f16 Rh[y] panel into
// LDS (with per-row padding for bank-conflict-free reads); each wave computes
// one x via 2 M-tiles x 8 N-tiles x 4 K-steps of v_wmma_f32_16x16x32_f16.
// ---------------------------------------------------------------------------
__global__ void __launch_bounds__(256)
maxsim_wmma(const _Float16* __restrict__ Lh,
            const _Float16* __restrict__ Rh,
            float* __restrict__ sim) {
  __shared__ _Float16 panel[LK * PSTRIDE];   // 34 KB, LDS offset 0

  const int wave = threadIdx.x >> 5;
  const int lane = threadIdx.x & 31;
  const int pair = blockIdx.x * 8 + wave;   // 0 .. 16383
  const int x    = pair & 127;              // varies within block
  const int y    = pair >> 7;               // uniform within block
  const int half = lane >> 4;               // 0: lanes 0-15, 1: lanes 16-31
  const int l16  = lane & 15;

  // Scalar (SGPR) leader test -> s_cbranch, NOT exec masking: tensor ops
  // ignore EXEC, so an exec-masked guard would issue 8 duplicate DMAs.
  const bool leader = (__builtin_amdgcn_readfirstlane(wave) == 0);

  // ---- Wave 0 kicks off the TDM: Rh[y] (128x128 f16, contiguous) -> LDS,
  // padding 4 dwords after every 64-dword row (PSTRIDE = 136 halves).
  if (leader) {
    const unsigned long long ga =
        (unsigned long long)(const void*)(Rh + (size_t)y * LK * DIM);
    uint32x4 g0;
    g0[0] = 1u;                                        // count=1, 2D, no gather
    g0[1] = 0u;                                        // lds_addr = 0
    g0[2] = (unsigned)(ga & 0xffffffffu);              // global_addr[31:0]
    g0[3] = (unsigned)((ga >> 32) & 0x01ffffffu)       // global_addr[56:32]
            | (2u << 30);                              // type=2 ("image")
    int32x8 g1;
    g1[0] = (int)0x07510000u;   // data_size=2B, pad_en, interval=64dw, pad=4dw
    g1[1] = (int)(128u << 16);  // tensor_dim0 = 128 (low 16 bits in [63:48])
    g1[2] = (int)(128u << 16);  // tensor_dim1 = 128
    g1[3] = (int)(128u << 16);  // tile_dim0   = 128
    g1[4] = (int)128;           // tile_dim1   = 128, tile_dim2 = 0
    g1[5] = (int)128;           // tensor_dim0_stride = 128
    g1[6] = 0;
    g1[7] = 0;
    const int32x4 gz4 = {0, 0, 0, 0};                  // groups 2/3 unused (2D)
    const int32x8 gz8 = {0, 0, 0, 0, 0, 0, 0, 0};
    __builtin_amdgcn_tensor_load_to_lds(g0, g1, gz4, gz4, gz8, 0);
  }

  // ---- Preload A fragments while the DMA runs: 2 M-tiles x 4 K-blocks
  // (A 16x32 map matching B: lanes 0-15 hold M=l16, K k0..k0+15;
  // lanes 16-31 hold K +16..+31).
  v16h a[2][4];
#pragma unroll
  for (int mt = 0; mt < 2; ++mt) {
    const _Float16* rowA = Lh + (size_t)(x * LQ + mt * 16 + l16) * DIM;
#pragma unroll
    for (int kb = 0; kb < 4; ++kb) {
      const int k0 = kb * 32 + half * 16;
      const v8h lo = *(const v8h*)(rowA + k0);
      const v8h hi = *(const v8h*)(rowA + k0 + 8);
      v16h f;
#pragma unroll
      for (int i = 0; i < 8; ++i) { f[i] = lo[i]; f[i + 8] = hi[i]; }
      a[mt][kb] = f;
    }
  }

  if (leader) __builtin_amdgcn_s_wait_tensorcnt(0);
  // Escape the panel's address into an opaque asm that may write memory.
  // `panel` is an internal LDS global with no visible store (the TDM writes
  // it behind the compiler's back); without this escape, GlobalOpt proves
  // "never stored" and folds every LDS load from it to undef — which is
  // exactly what deleted 56 of the 64 WMMAs in the previous rounds.
  asm volatile("" : : "v"(&panel[0]) : "memory");
  __syncthreads();

  // Running per-row maxes. C/D layout: lane -> N (l16), VGPR r -> M row
  // (r for lanes 0-15, r+8 for lanes 16-31).
  float rm0[8], rm1[8];
#pragma unroll
  for (int r = 0; r < 8; ++r) { rm0[r] = -3.0e38f; rm1[r] = -3.0e38f; }

  // B fragment (32x16) from LDS: lane l16 holds column N = nt*16+l16,
  // contiguous K run of 16 halves starting at kb*32 (+16 for lanes 16-31).
  const _Float16* pB0 = panel + (size_t)l16 * PSTRIDE + half * 16;

#pragma unroll
  for (int nt = 0; nt < 8; ++nt) {
    v8f acc0 = {};
    v8f acc1 = {};
    const _Float16* pB = pB0 + (size_t)nt * 16 * PSTRIDE;
#pragma unroll
    for (int kb = 0; kb < 4; ++kb) {
      const v16h b = *(const v16h*)(pB + kb * 32);
      acc0 = __builtin_amdgcn_wmma_f32_16x16x32_f16(
          false, a[0][kb], false, b, (short)0, acc0, false, false);
      acc1 = __builtin_amdgcn_wmma_f32_16x16x32_f16(
          false, a[1][kb], false, b, (short)0, acc1, false, false);
    }
#pragma unroll
    for (int r = 0; r < 8; ++r) {
      rm0[r] = fmaxf(rm0[r], acc0[r]);
      rm1[r] = fmaxf(rm1[r], acc1[r]);
    }
  }

  // Reduce: max over N lives across the 16-lane half; then sum the 32 rows.
  float s = 0.0f;
#pragma unroll
  for (int r = 0; r < 8; ++r) s += halfmax16(rm0[r]);  // rows 0-7 / 8-15
#pragma unroll
  for (int r = 0; r < 8; ++r) s += halfmax16(rm1[r]);  // rows 16-23 / 24-31
  s += __shfl_xor(s, 16);                               // combine halves
  if (lane == 0) sim[x * BATCH + y] = s;
}

// ---------------------------------------------------------------------------
// Kernel 3: loss = -mean_x( scale*sim[x,pos[x]] - logsumexp_y(scale*sim[x,y]) )
// ---------------------------------------------------------------------------
__global__ void __launch_bounds__(128)
ce_loss(const float* __restrict__ sim,
        const float* __restrict__ logit_scale,
        const long long* __restrict__ pos,
        float* __restrict__ out) {
  __shared__ float red[BATCH];
  const int x = threadIdx.x;
  const float s = expf(logit_scale[0]);
  const float* row = sim + x * BATCH;

  float m = -3.0e38f;
  for (int yy = 0; yy < BATCH; ++yy) m = fmaxf(m, s * row[yy]);
  float se = 0.0f;
  for (int yy = 0; yy < BATCH; ++yy) se += expf(s * row[yy] - m);
  const float lse = m + logf(se);
  const int p = (int)pos[x];
  red[x] = lse - s * row[p];
  __syncthreads();

  for (int stride = BATCH / 2; stride > 0; stride >>= 1) {
    if (x < stride) red[x] += red[x + stride];
    __syncthreads();
  }
  if (x == 0) out[0] = red[0] / (float)BATCH;
}

// ---------------------------------------------------------------------------
extern "C" void kernel_launch(void* const* d_in, const int* in_sizes, int n_in,
                              void* d_out, int out_size, void* d_ws, size_t ws_size,
                              hipStream_t stream) {
  (void)in_sizes; (void)n_in; (void)out_size; (void)ws_size;
  const float*     left        = (const float*)d_in[0];      // [128,32,128] f32
  const float*     right       = (const float*)d_in[1];      // [128,128,128] f32
  const float*     logit_scale = (const float*)d_in[2];      // scalar f32
  const long long* pos         = (const long long*)d_in[3];  // [128] i64
  float*           out         = (float*)d_out;              // scalar f32

  _Float16* Lh  = (_Float16*)d_ws;                           // 1 MB
  _Float16* Rh  = Lh + (size_t)BATCH * LQ * DIM;             // 4 MB
  float*    sim = (float*)(Rh + (size_t)BATCH * LK * DIM);   // 64 KB

  normalize_to_f16<<<(BATCH * LQ + BATCH * LK) / 8, 256, 0, stream>>>(
      left, right, Lh, Rh);
  maxsim_wmma<<<(BATCH * BATCH) / 8, 256, 0, stream>>>(Lh, Rh, sim);
  ce_loss<<<1, BATCH, 0, stream>>>(sim, logit_scale, pos, out);
}
